// LAHead18a_61108794687938
// MI455X (gfx1250) — matrix operand order repaired and checked by reading, not compiled, not run on hardware
//
#include <hip/hip_runtime.h>
#include <hip/hip_bf16.h>
#include <math.h>
#include <stdint.h>

// ---- dims ----
#define BB 2
#define HC 96
#define WC 96
#define HF 384
#define WF 384
#define CINC 256
#define CH 64
#define COUT 19
#define KTOT 2304          // 256*9
#define KV 320             // 256 + 64

typedef __bf16 v16bf __attribute__((ext_vector_type(16)));
typedef __bf16 v8bf  __attribute__((ext_vector_type(8)));
typedef float  v8f   __attribute__((ext_vector_type(8)));

static __device__ __forceinline__ v16bf cat16(v8bf lo, v8bf hi) {
  return __builtin_shufflevector(lo, hi, 0, 1, 2, 3, 4, 5, 6, 7,
                                 8, 9, 10, 11, 12, 13, 14, 15);
}

// Async global->LDS copy (CDNA5, ASYNCcnt-tracked). GVS mode: SGPR base +
// per-lane 32-bit byte offset; VGPR operand 0 carries the LDS byte address.
static __device__ __forceinline__ void async_ld32(void* lds_ptr,
                                                  const void* gbase,
                                                  uint32_t byte_off) {
  uint32_t la = (uint32_t)(uintptr_t)lds_ptr;
  asm volatile("global_load_async_to_lds_b32 %0, %1, %2 offset:0"
               :
               : "v"(la), "v"(byte_off), "s"((uint64_t)(uintptr_t)gbase)
               : "memory");
}
static __device__ __forceinline__ void wait_async0() {
  asm volatile("s_wait_asynccnt 0x0" ::: "memory");
}

// ---------------- Kernel 0: weight conversion fp32 -> bf16 ----------------
__global__ __launch_bounds__(256) void prep_weights(
    const float* __restrict__ wec, const float* __restrict__ wv,
    __bf16* __restrict__ wec_bf, __bf16* __restrict__ wv_bf) {
  int i = blockIdx.x * 256 + threadIdx.x;
  if (i < CH * KTOT) wec_bf[i] = (__bf16)wec[i];
  if (i < 32 * KV) {
    int co = i / KV, ci = i - co * KV;
    wv_bf[i] = (co < COUT) ? (__bf16)wv[co * KV + ci] : (__bf16)0.0f;
  }
}

// ---------------- Kernel 1: ec = relu(conv3x3(coarse, w_ec)) via WMMA -----
// Implicit GEMM, M=64 (4 waves x 16), K=2304, N = 4x4 pixel tile.
// im2col panel [n][k] in LDS makes every WMMA fragment a contiguous load:
//   A: two 16B runs of w_ec row;  B: two 16B runs of im2col row.
__global__ __launch_bounds__(128) void ec_conv_wmma(
    const float* __restrict__ coarse, const __bf16* __restrict__ wec_bf,
    const float* __restrict__ b_ec, float* __restrict__ ec) {
  __shared__ __bf16 patch[CINC * 36];       // 6x6 halo patch, 256 ch (18 KB)
  __shared__ __bf16 im2col[16 * KTOT];      // [n][k] panel (72 KB)
  const int b  = blockIdx.z;
  const int h0 = blockIdx.y * 4, w0 = blockIdx.x * 4;
  const int tid = threadIdx.x;

  for (int idx = tid; idx < CINC * 36; idx += 128) {
    int ci = idx / 36, pos = idx - ci * 36;
    int y = pos / 6, x = pos - y * 6;
    int gh = h0 - 1 + y, gw = w0 - 1 + x;
    float v = 0.f;
    if (gh >= 0 && gh < HC && gw >= 0 && gw < WC)
      v = coarse[((b * CINC + ci) * HC + gh) * WC + gw];
    patch[idx] = (__bf16)v;
  }
  __syncthreads();

  // build im2col: element (n,k) = patch[ci][py+dh][px+dw], k = ci*9 + r
  for (int n2 = 0; n2 < 16; ++n2) {
    int py2 = n2 >> 2, px2 = n2 & 3;
    for (int k = tid; k < KTOT; k += 128) {
      int ci = k / 9, r = k - ci * 9;
      int dh = r / 3, dw = r - dh * 3;
      im2col[n2 * KTOT + k] = patch[ci * 36 + (py2 + dh) * 6 + (px2 + dw)];
    }
  }
  __syncthreads();

  const int wave  = tid >> 5, lane = tid & 31;
  const int khalf = lane >> 4;               // K-half select per ISA layout
  const int mglob = wave * 16 + (lane & 15);
  const int n  = lane & 15;                  // pixel within 4x4 tile
  const int py = n >> 2, px = n & 3;

  const __bf16* arow = wec_bf + (size_t)mglob * KTOT;
  const __bf16* brow = &im2col[n * KTOT];
  __builtin_prefetch(arow, 0, 3);            // global_prefetch_b8

  v8f acc = {};
  for (int kc = 0; kc < KTOT; kc += 32) {
    v8bf a0 = *(const v8bf*)(arow + kc + khalf * 8);
    v8bf a1 = *(const v8bf*)(arow + kc + 16 + khalf * 8);
    v8bf b0 = *(const v8bf*)(brow + kc + khalf * 8);
    v8bf b1 = *(const v8bf*)(brow + kc + 16 + khalf * 8);
    acc = __builtin_amdgcn_wmma_f32_16x16x32_bf16(false, cat16(a0, a1),
                                                  false, cat16(b0, b1),
                                                  (short)0, acc, false, false);
  }

#pragma unroll
  for (int j = 0; j < 8; ++j) {              // C: M = j + 8*khalf, N = lane&15
    int m = wave * 16 + j + khalf * 8;
    float v = acc[j] + b_ec[m];
    v = v > 0.f ? v : 0.f;
    ec[((b * CH + m) * HC + h0 + py) * WC + (w0 + px)] = v;
  }
}

// ---------------- Kernel 2: fused edge branch + xq over the fine grid -----
__global__ __launch_bounds__(256) void fine_fused(
    const float* __restrict__ raw, const float* __restrict__ ecbuf,
    const float* __restrict__ w_ef, const float* __restrict__ b_ef,
    const float* __restrict__ w_eh,
    const float* __restrict__ w_q, const float* __restrict__ b_q,
    float* __restrict__ xq, float* __restrict__ ne_out) {
  __shared__ float s_wef[27], s_weh[65], s_wq[192], s_bq[64];
  int t = threadIdx.x;
  if (t < 27)  s_wef[t] = w_ef[t];
  if (t < 65)  s_weh[t] = w_eh[t];
  if (t < 192) s_wq[t]  = w_q[t];
  if (t < 64)  s_bq[t]  = b_q[t];
  __syncthreads();

  int idx = blockIdx.x * 256 + t;            // exactly B*HF*WF threads
  int b = idx / (HF * WF);
  int p = idx - b * (HF * WF);
  int h = p / WF, w = p - h * WF;

  // ef = relu(conv3x3(raw; 3ch) + b_ef)
  float ef = b_ef[0];
#pragma unroll
  for (int c = 0; c < 3; ++c)
#pragma unroll
    for (int dh = -1; dh <= 1; ++dh)
#pragma unroll
      for (int dw = -1; dw <= 1; ++dw) {
        int hh = h + dh, ww = w + dw;
        if (hh >= 0 && hh < HF && ww >= 0 && ww < WF)
          ef += s_wef[(c * 3 + dh + 1) * 3 + (dw + 1)] *
                raw[((b * 3 + c) * HF + hh) * WF + ww];
      }
  ef = ef > 0.f ? ef : 0.f;

  float r0 = raw[((b * 3 + 0) * HF + h) * WF + w];
  float r1 = raw[((b * 3 + 1) * HF + h) * WF + w];
  float r2 = raw[((b * 3 + 2) * HF + h) * WF + w];

  int hc = h >> 2, wc = w >> 2;
  float el = s_weh[64] * ef;                 // concat: [ec_up (0..63), ef (64)]
  for (int c = 0; c < CH; ++c) {
    el += s_weh[c] * ecbuf[((b * CH + c) * HC + hc) * WC + wc];
    float q = s_wq[c * 3] * r0 + s_wq[c * 3 + 1] * r1 + s_wq[c * 3 + 2] * r2 + s_bq[c];
    xq[((b * CH + c) * HF + h) * WF + w] = q;
  }
  ne_out[b * (HF * WF) + p] = 1.f / (1.f + expf(-el));
}

// ---------------- Kernel 3: xv = conv1x1([coarse_up ; xq]) via WMMA -------
// One 4x4 fine tile == one coarse pixel; B-panel [n][k] in LDS (10 KB) gives
// contiguous fragment loads just like kernel 1.
__global__ __launch_bounds__(64) void xv_wmma(
    const float* __restrict__ coarse, const float* __restrict__ xq,
    const __bf16* __restrict__ wv_bf, const float* __restrict__ b_v,
    float* __restrict__ xv) {
  __shared__ __bf16 s_c[CINC];
  __shared__ __bf16 bpanel[16 * KV];
  int b  = blockIdx.z;
  int wc = blockIdx.x, hc = blockIdx.y;
  int h0 = hc * 4, w0 = wc * 4;
  int tid = threadIdx.x;

  for (int i = tid; i < CINC; i += 64)
    s_c[i] = (__bf16)coarse[((b * CINC + i) * HC + hc) * WC + wc];
  for (int i = tid; i < CH * 16; i += 64) {
    int c = i >> 4, nn = i & 15;
    int py = nn >> 2, px = nn & 3;
    bpanel[nn * KV + CINC + c] =
        (__bf16)xq[((b * CH + c) * HF + h0 + py) * WF + (w0 + px)];
  }
  __syncthreads();
  for (int i = tid; i < 16 * CINC; i += 64) {
    int nn = i >> 8, k = i & 255;            // coarse rows broadcast to all n
    bpanel[nn * KV + k] = s_c[k];
  }
  __syncthreads();

  int wave = tid >> 5, lane = tid & 31;      // wave = M-tile (rows 0-15/16-31)
  int khalf = lane >> 4;
  int co_base = wave * 16;
  int n = lane & 15, py = n >> 2, px = n & 3;

  const __bf16* arow = wv_bf + (co_base + (lane & 15)) * KV;
  const __bf16* brow = &bpanel[n * KV];
  __builtin_prefetch(arow, 0, 3);

  v8f acc = {};
  for (int kc = 0; kc < KV; kc += 32) {
    v8bf a0 = *(const v8bf*)(arow + kc + khalf * 8);
    v8bf a1 = *(const v8bf*)(arow + kc + 16 + khalf * 8);
    v8bf b0 = *(const v8bf*)(brow + kc + khalf * 8);
    v8bf b1 = *(const v8bf*)(brow + kc + 16 + khalf * 8);
    acc = __builtin_amdgcn_wmma_f32_16x16x32_bf16(false, cat16(a0, a1),
                                                  false, cat16(b0, b1),
                                                  (short)0, acc, false, false);
  }

#pragma unroll
  for (int j = 0; j < 8; ++j) {
    int co = co_base + j + khalf * 8;
    if (co < COUT)
      xv[((b * COUT + co) * HF + h0 + py) * WF + (w0 + px)] = acc[j] + b_v[co];
  }
}

// ---------------- Kernel 4: 3x3 local attention ---------------------------
// score_t(p) = ne(p+o) * <q(p), q(p+o)>, zero-padded (matches the reference's
// zero border scores inside softmax). Halo staged with async global->LDS
// copies (ASYNCcnt); out-of-bounds cells zeroed via DS stores (disjoint).
__global__ __launch_bounds__(64) void local_attn(
    const float* __restrict__ xq, const float* __restrict__ xv,
    const float* __restrict__ ne, float* __restrict__ y) {
  __shared__ float s_q[CH][100];
  __shared__ float s_v[COUT][100];
  __shared__ float s_ne[100];
  int b  = blockIdx.z;
  int h0 = blockIdx.y * 8, w0 = blockIdx.x * 8;
  int tid = threadIdx.x;

  for (int i = tid; i < 100; i += 64) {
    int ly = i / 10, lx = i - ly * 10;
    int gh = h0 - 1 + ly, gw = w0 - 1 + lx;
    if (gh >= 0 && gh < HF && gw >= 0 && gw < WF) {
      uint32_t pix = (uint32_t)(gh * WF + gw);
      async_ld32(&s_ne[i], ne, 4u * (uint32_t)(b * (HF * WF)) + 4u * pix);
      for (int c = 0; c < CH; ++c)
        async_ld32(&s_q[c][i], xq,
                   4u * ((uint32_t)((b * CH + c) * (HF * WF)) + pix));
      for (int c = 0; c < COUT; ++c)
        async_ld32(&s_v[c][i], xv,
                   4u * ((uint32_t)((b * COUT + c) * (HF * WF)) + pix));
    } else {
      s_ne[i] = 0.f;
      for (int c = 0; c < CH; ++c)   s_q[c][i] = 0.f;
      for (int c = 0; c < COUT; ++c) s_v[c][i] = 0.f;
    }
  }
  wait_async0();
  __syncthreads();

  int py = tid >> 3, px = tid & 7;
  int lc = (py + 1) * 10 + (px + 1);

  float sc[9];
#pragma unroll
  for (int t = 0; t < 9; ++t) {
    int dh = t / 3 - 1, dw = t % 3 - 1;
    int ln = (py + 1 + dh) * 10 + (px + 1 + dw);
    float dot = 0.f;
    for (int c = 0; c < CH; ++c) dot += s_q[c][lc] * s_q[c][ln];
    sc[t] = dot * s_ne[ln];
  }
  float mx = sc[0];
#pragma unroll
  for (int t = 1; t < 9; ++t) mx = fmaxf(mx, sc[t]);
  float se = 0.f;
#pragma unroll
  for (int t = 0; t < 9; ++t) { sc[t] = expf(sc[t] - mx); se += sc[t]; }
  float inv = 1.f / se;

  for (int c = 0; c < COUT; ++c) {
    float acc = 0.f;
#pragma unroll
    for (int t = 0; t < 9; ++t) {
      int dh = t / 3 - 1, dw = t % 3 - 1;
      acc += sc[t] * s_v[c][(py + 1 + dh) * 10 + (px + 1 + dw)];
    }
    y[((b * COUT + c) * HF + h0 + py) * WF + (w0 + px)] = acc * inv;
  }
}

// ---------------- launcher -------------------------------------------------
extern "C" void kernel_launch(void* const* d_in, const int* in_sizes, int n_in,
                              void* d_out, int out_size, void* d_ws, size_t ws_size,
                              hipStream_t stream) {
  const float* coarse = (const float*)d_in[0];
  const float* raw    = (const float*)d_in[1];
  const float* w_ef   = (const float*)d_in[2];
  const float* b_ef   = (const float*)d_in[3];
  const float* w_ec   = (const float*)d_in[4];
  const float* b_ec   = (const float*)d_in[5];
  const float* w_eh   = (const float*)d_in[6];
  const float* w_q    = (const float*)d_in[7];
  const float* b_q    = (const float*)d_in[8];
  const float* w_v    = (const float*)d_in[9];
  const float* b_v    = (const float*)d_in[10];

  // workspace layout (bytes); total ~103 MB
  char* ws = (char*)d_ws;
  __bf16* wec_bf = (__bf16*)(ws + 0);           //   294,912 B (64x2304 bf16)
  __bf16* wv_bf  = (__bf16*)(ws + 294912);      //    20,480 B (32x320 bf16, padded)
  float*  ec     = (float*)(ws + 315392);       // 4,718,592 B (B,64,96,96)
  float*  xq     = (float*)(ws + 5033984);      // 75,497,472 B (B,64,384,384)
  float*  xv     = (float*)(ws + 80531456);     // 22,413,312 B (B,19,384,384)

  float* y_out  = (float*)d_out;                       // (B,19,384,384)
  float* ne_out = y_out + (size_t)BB * COUT * HF * WF; // (B,1,384,384)

  prep_weights<<<576, 256, 0, stream>>>(w_ec, w_v, wec_bf, wv_bf);
  ec_conv_wmma<<<dim3(WC / 4, HC / 4, BB), 128, 0, stream>>>(coarse, wec_bf, b_ec, ec);
  fine_fused<<<(BB * HF * WF) / 256, 256, 0, stream>>>(raw, ec, w_ef, b_ef, w_eh,
                                                       w_q, b_q, xq, ne_out);
  xv_wmma<<<dim3(WC, HC, BB), 64, 0, stream>>>(coarse, xq, wv_bf, b_v, xv);
  local_attn<<<dim3(WF / 8, HF / 8, BB), 64, 0, stream>>>(xq, xv, ne_out, y_out);
}
